// Attention_10393820856760
// MI455X (gfx1250) — compile-verified, matrix operand order checked by reading, not run on hardware
//
#include <hip/hip_runtime.h>

// ---------------------------------------------------------------------------
// Attention decoder (CRNN-style) for MI455X / gfx1250.
//  - GEMMs: v_wmma_f32_16x16x32_bf16 (f32 accum); weights pre-packed bf16.
//  - B tiles staged via the Tensor Data Mover (tensor_load_to_lds +
//    s_wait_tensorcnt) when the builtin is available; vector-copy fallback.
//  - Software-pipelined K loop with double-buffered LDS tiles: next tile's
//    global loads + TDM issue before the current tile's WMMAs, conversion &
//    LDS stores after them, one barrier per K-step.
//  - All dims padded so staging/stores need no bounds checks: M%128==0,
//    N%64==0, K%32==0 everywhere.
//  - Attention score/softmax/context is VALU (tanh-bound), one WG per row.
//  - One-hot @ W_ih is a column gather folded into the LSTM pointwise kernel.
// ---------------------------------------------------------------------------

typedef unsigned short u16;
typedef __bf16 bf16_t;
typedef bf16_t v16bf  __attribute__((ext_vector_type(16)));
typedef u16    v4u16  __attribute__((ext_vector_type(4)));
typedef u16    v8u16  __attribute__((ext_vector_type(8)));
typedef u16    v16u16 __attribute__((ext_vector_type(16)));
typedef float  v4f    __attribute__((ext_vector_type(4)));
typedef float  v8f    __attribute__((ext_vector_type(8)));

#define B_SZ   512
#define T_SZ   128
#define IN_SZ  256
#define HID    256
#define NC     100
#define NPAD   128            // NC padded to multiple of 64 (GEMM N tile)
#define STEPS  26
#define BLANK  3
#define WIH_W  (IN_SZ + NC)   // 356, W_ih row width
#define KCAT   (IN_SZ + HID)  // 512, fused gates GEMM K
#define LDSROW 40             // u16 per LDS row: 32 data + 8 pad (4 dwords)

__device__ __forceinline__ u16 f2bf(float f) {
  unsigned u = __builtin_bit_cast(unsigned, f);
  unsigned r = u + 0x7FFFu + ((u >> 16) & 1u);   // round-to-nearest-even
  return (u16)(r >> 16);
}

// Load one 16x32 bf16 fragment from LDS per the CDNA5 16-bit A/B layout:
// lane L holds row (L&15); elements 0..7 = K {ks..ks+7},
// elements 8..15 = K {16+ks..23+ks}, ks = (L>=16 ? 8 : 0).
__device__ __forceinline__ v16bf load_frag(const u16* rowp, int ks) {
  v8u16 lo = *reinterpret_cast<const v8u16*>(rowp + ks);
  v8u16 hi = *reinterpret_cast<const v8u16*>(rowp + 16 + ks);
  union { v16u16 u; v8u16 h[2]; } t;
  t.h[0] = lo; t.h[1] = hi;
  return __builtin_bit_cast(v16bf, t.u);
}

// ---------------------------------------------------------------------------
// Tensor Data Mover: DMA one 64x32 bf16 tile (row stride ldb elements) from
// global into LDS with a 4-dword pad after every 16 dwords (= one 32-elem
// row), matching the Bs[64][LDSROW] layout. 2D descriptor, groups 2/3 zero.
// ---------------------------------------------------------------------------
#if __has_builtin(__builtin_amdgcn_tensor_load_to_lds) && \
    __has_builtin(__builtin_amdgcn_s_wait_tensorcnt)
#define HAVE_TDM 1
typedef unsigned int u32x4 __attribute__((ext_vector_type(4)));
typedef int          i32x4 __attribute__((ext_vector_type(4)));
typedef int          i32x8 __attribute__((ext_vector_type(8)));

__device__ __forceinline__ void tdm_load_b_tile(const u16* gsrc,
                                                unsigned lds_addr, int ldb) {
  unsigned long long ga = (unsigned long long)(__UINTPTR_TYPE__)gsrc;
  u32x4 g0;
  g0[0] = 1u;                                   // count=1, user mode, no gather
  g0[1] = lds_addr;                             // LDS byte address of tile
  g0[2] = (unsigned)(ga & 0xffffffffu);         // global_addr[31:0]
  g0[3] = (unsigned)((ga >> 32) & 0x01ffffffu)  // global_addr[56:32]
          | (2u << 30);                         // type=2 ("image")
  i32x8 g1;
  g1[0] = (int)((1u << 16)      // data_size = 2 bytes
              | (1u << 20)      // pad_enable
              | (3u << 22)      // pad_interval: 16 dwords (one 32-elem row)
              | (3u << 25));    // pad_amount:   4 dwords (8 u16)
  g1[1] = (int)(((unsigned)ldb & 0xffffu) << 16);              // tensor_dim0 lo
  g1[2] = (int)((((unsigned)ldb >> 16) & 0xffffu) | (64u << 16)); // dim0 hi | tensor_dim1=64
  g1[3] = (int)(32u << 16);     // tensor_dim1 hi=0 | tile_dim0=32
  g1[4] = 64;                   // tile_dim1=64 | tile_dim2=0
  g1[5] = ldb;                  // tensor_dim0_stride (elements)
  g1[6] = 0;
  g1[7] = 0;
  i32x4 z4 = {0, 0, 0, 0};
#if __has_include(<hip/amd_detail/amd_gfx1250_TDM.h>)
  i32x8 z8 = {0, 0, 0, 0, 0, 0, 0, 0};
  __builtin_amdgcn_tensor_load_to_lds(g0, g1, z4, z4, z8, 0);   // clang-23 form
#else
  __builtin_amdgcn_tensor_load_to_lds(g0, g1, z4, z4, 0);       // ROCm 7.2 form
#endif
}
#endif

// ---------------------------------------------------------------------------
// C[M,N] = A[M,K](f32) * Bw[N,K](bf16 row-major, K contiguous)^T (+ bias[n]).
// Block tile 128x64, 8 waves (4M x 2N), each wave 2x2 16x16x32 WMMA tiles.
// Double-buffered, software-pipelined K loop (one barrier per K-step).
// REQUIRES: M%128==0, N%64==0, K%32==0 (all call sites padded accordingly).
// ---------------------------------------------------------------------------
__global__ __launch_bounds__(256)
void gemm_wmma_bf16(const float* __restrict__ A, int lda,
                    const u16* __restrict__ Bw, int ldb,
                    float* __restrict__ C, int ldc,
                    const float* __restrict__ bias,
                    int M, int N, int K) {
  __shared__ __align__(16) u16 As[2][128 * LDSROW];
  __shared__ __align__(16) u16 Bs[2][64 * LDSROW];

  const int tid  = threadIdx.x;
  const int lane = tid & 31;
  const int wave = tid >> 5;
  const int m0 = blockIdx.y * 128;
  const int n0 = blockIdx.x * 64;
  const int wm = (wave & 3) * 32;
  const int wn = (wave >> 2) * 32;

  // A staging geometry: 4 chunks of 4 floats per thread (128x32 tile)
  int ar[4], ac4[4];
#pragma unroll
  for (int i = 0; i < 4; ++i) {
    int ch = tid + i * 256;
    ar[i] = ch >> 3;
    ac4[i] = (ch & 7) * 4;
  }

  v8f acc[2][2];
#pragma unroll
  for (int a = 0; a < 2; ++a)
#pragma unroll
    for (int b = 0; b < 2; ++b)
#pragma unroll
      for (int e = 0; e < 8; ++e) acc[a][b][e] = 0.f;

  const int ks   = (lane & 16) ? 8 : 0;
  const int frow = lane & 15;

  v4f t[4];
#ifndef HAVE_TDM
  const int br = tid >> 2, bc8 = (tid & 3) * 8;
  v8u16 bt;
#endif

  // ---- prologue: stage tile 0 into buffer 0 -------------------------------
#pragma unroll
  for (int i = 0; i < 4; ++i)
    t[i] = *reinterpret_cast<const v4f*>(&A[(long)(m0 + ar[i]) * lda + ac4[i]]);
#ifdef HAVE_TDM
  if (wave == 0)
    tdm_load_b_tile(Bw + (long)n0 * ldb,
                    (unsigned)(__UINTPTR_TYPE__)&Bs[0][0], ldb);
#else
  bt = *reinterpret_cast<const v8u16*>(&Bw[(long)(n0 + br) * ldb + bc8]);
#endif
#pragma unroll
  for (int i = 0; i < 4; ++i) {
    v4u16 o;
    o[0] = f2bf(t[i][0]); o[1] = f2bf(t[i][1]);
    o[2] = f2bf(t[i][2]); o[3] = f2bf(t[i][3]);
    *reinterpret_cast<v4u16*>(&As[0][ar[i] * LDSROW + ac4[i]]) = o;
  }
#ifndef HAVE_TDM
  *reinterpret_cast<v8u16*>(&Bs[0][br * LDSROW + bc8]) = bt;
#endif
#ifdef HAVE_TDM
  if (wave == 0) __builtin_amdgcn_s_wait_tensorcnt(0);
#endif
  __syncthreads();

  // ---- pipelined main loop ------------------------------------------------
  int buf = 0;
  for (int kt = 0; kt < K; kt += 32, buf ^= 1) {
    const int  nbuf = buf ^ 1;
    const bool more = (kt + 32) < K;

    if (more) {  // issue next tile's loads before computing (latency hiding)
#pragma unroll
      for (int i = 0; i < 4; ++i)
        t[i] = *reinterpret_cast<const v4f*>(
            &A[(long)(m0 + ar[i]) * lda + kt + 32 + ac4[i]]);
#ifdef HAVE_TDM
      if (wave == 0)
        tdm_load_b_tile(Bw + (long)n0 * ldb + kt + 32,
                        (unsigned)(__UINTPTR_TYPE__)&Bs[nbuf][0], ldb);
#else
      bt = *reinterpret_cast<const v8u16*>(
          &Bw[(long)(n0 + br) * ldb + kt + 32 + bc8]);
#endif
    }

    // compute current tile
    v16bf fa[2], fb[2];
#pragma unroll
    for (int q = 0; q < 2; ++q)
      fa[q] = load_frag(&As[buf][(wm + q * 16 + frow) * LDSROW], ks);
#pragma unroll
    for (int q = 0; q < 2; ++q)
      fb[q] = load_frag(&Bs[buf][(wn + q * 16 + frow) * LDSROW], ks);
#pragma unroll
    for (int ti = 0; ti < 2; ++ti)
#pragma unroll
      for (int tj = 0; tj < 2; ++tj)
        acc[ti][tj] = __builtin_amdgcn_wmma_f32_16x16x32_bf16(
            false, fa[ti], false, fb[tj], (short)0, acc[ti][tj], false, false);

    if (more) {  // convert + publish next tile (loadcnt wait lands here)
#pragma unroll
      for (int i = 0; i < 4; ++i) {
        v4u16 o;
        o[0] = f2bf(t[i][0]); o[1] = f2bf(t[i][1]);
        o[2] = f2bf(t[i][2]); o[3] = f2bf(t[i][3]);
        *reinterpret_cast<v4u16*>(&As[nbuf][ar[i] * LDSROW + ac4[i]]) = o;
      }
#ifndef HAVE_TDM
      *reinterpret_cast<v8u16*>(&Bs[nbuf][br * LDSROW + bc8]) = bt;
#endif
#ifdef HAVE_TDM
      if (wave == 0) __builtin_amdgcn_s_wait_tensorcnt(0);
#endif
    }
    __syncthreads();
  }

  // D layout: VGPR v, lanes 0-15 -> M=v, N=lane; lanes 16-31 -> M=v+8, N=lane-16
  const int rofs = (lane & 16) ? 8 : 0;
  const int ncol = lane & 15;
#pragma unroll
  for (int ti = 0; ti < 2; ++ti)
#pragma unroll
    for (int tj = 0; tj < 2; ++tj) {
      int n = n0 + wn + tj * 16 + ncol;
      float bv = bias ? bias[n] : 0.f;
#pragma unroll
      for (int v = 0; v < 8; ++v) {
        int m = m0 + wm + ti * 16 + v + rofs;
        C[(long)m * ldc + n] = acc[ti][tj][v] + bv;
      }
    }
}

// ---------------------------------------------------------------------------
// Per batch row b: e[t] = sum_h tanh(proj_H[b,t,h]+proj_h[b,h])*w_score[h];
// alpha = softmax_t(e); context[b,:] = sum_t alpha[t]*batch_H[b,t,:]
// context written into xcat[b, 0:256] (gates GEMM input).
// ---------------------------------------------------------------------------
__global__ __launch_bounds__(256)
void attention_ctx(const float* __restrict__ proj_H,
                   const float* __restrict__ proj_h,
                   const float* __restrict__ batch_H,
                   const float* __restrict__ w_score,
                   float* __restrict__ xcat) {
  __shared__ float ph[HID];
  __shared__ float wsc[HID];
  __shared__ float e_s[T_SZ];
  __shared__ float red[T_SZ];
  __shared__ float s_max, s_sum;

  const int b = blockIdx.x, tid = threadIdx.x;
  ph[tid]  = proj_h[b * HID + tid];
  wsc[tid] = w_score[tid];
  __syncthreads();

  const int lane = tid & 31, wave = tid >> 5;
  for (int t = wave; t < T_SZ; t += 8) {
    const float* row = proj_H + ((long)b * T_SZ + t) * HID;
    float sum = 0.f;
    for (int h = lane; h < HID; h += 32)
      sum += tanhf(row[h] + ph[h]) * wsc[h];
#pragma unroll
    for (int o = 16; o > 0; o >>= 1) sum += __shfl_down(sum, o, 32);
    if (lane == 0) e_s[t] = sum;
  }
  __syncthreads();

  if (tid < T_SZ) red[tid] = e_s[tid];
  __syncthreads();
  for (int s = T_SZ / 2; s > 0; s >>= 1) {
    if (tid < s) red[tid] = fmaxf(red[tid], red[tid + s]);
    __syncthreads();
  }
  if (tid == 0) s_max = red[0];
  __syncthreads();
  if (tid < T_SZ) { float p = __expf(e_s[tid] - s_max); e_s[tid] = p; red[tid] = p; }
  __syncthreads();
  for (int s = T_SZ / 2; s > 0; s >>= 1) {
    if (tid < s) red[tid] += red[tid + s];
    __syncthreads();
  }
  if (tid == 0) s_sum = red[0];
  __syncthreads();

  const float inv = 1.f / s_sum;
  const float* bh = batch_H + (long)b * T_SZ * IN_SZ;
  float ctx = 0.f;
  for (int t = 0; t < T_SZ; ++t)          // per fixed t, lanes coalesce
    ctx += e_s[t] * bh[t * IN_SZ + tid];
  xcat[b * KCAT + tid] = ctx * inv;
}

// ---------------------------------------------------------------------------
// LSTM pointwise; one-hot contribution = W_ih column gather.
// Writes h into xcat[b, 256:512] (next step's GEMM input) and out_hid.
// ---------------------------------------------------------------------------
__global__ __launch_bounds__(256)
void lstm_cell(const float* __restrict__ gates,
               const float* __restrict__ W_ih,
               const int* __restrict__ text, int step,
               float* __restrict__ cbuf,
               float* __restrict__ xcat,
               float* __restrict__ out_hid) {
  const int b = blockIdx.x, j = threadIdx.x;
  const int col = IN_SZ + text[b * STEPS + step];
  const float* g = gates + (long)b * (4 * HID);

  float gi = g[j]           + W_ih[(long)(0 * HID + j) * WIH_W + col];
  float gf = g[HID + j]     + W_ih[(long)(1 * HID + j) * WIH_W + col];
  float gg = g[2 * HID + j] + W_ih[(long)(2 * HID + j) * WIH_W + col];
  float go = g[3 * HID + j] + W_ih[(long)(3 * HID + j) * WIH_W + col];

  float cv = cbuf[b * HID + j];
  float si = 1.f / (1.f + __expf(-gi));
  float sf = 1.f / (1.f + __expf(-gf));
  float so = 1.f / (1.f + __expf(-go));
  float cn = sf * cv + si * tanhf(gg);
  float hn = so * tanhf(cn);

  cbuf[b * HID + j] = cn;
  xcat[b * KCAT + IN_SZ + j] = hn;
  out_hid[((long)b * STEPS + step) * HID + j] = hn;
}

// ---------------------------------------------------------------------------
// One-shot weight packing (f32 -> bf16) + bias prep, re-done every launch.
// ---------------------------------------------------------------------------
__global__ void convert_weights(const float* __restrict__ W_i2h,
                                const float* __restrict__ W_h2h,
                                const float* __restrict__ W_ih,
                                const float* __restrict__ W_hh,
                                const float* __restrict__ W_gen,
                                const float* __restrict__ b_ih,
                                const float* __restrict__ b_hh,
                                const float* __restrict__ b_gen,
                                u16* wi2h_bf, u16* wh2h_bf, u16* wcat_bf,
                                u16* wgen_bf, float* bsum, float* bgen_pad,
                                int total) {
  int idx = blockIdx.x * blockDim.x + threadIdx.x;
  if (idx >= total) return;
  if (idx < HID * IN_SZ) { wi2h_bf[idx] = f2bf(W_i2h[idx]); return; }
  idx -= HID * IN_SZ;
  if (idx < HID * HID) { wh2h_bf[idx] = f2bf(W_h2h[idx]); return; }
  idx -= HID * HID;
  if (idx < 4 * HID * KCAT) {   // fused [W_ih[:, :256] ; W_hh] : 1024 x 512
    int j = idx / KCAT, k = idx % KCAT;
    float v = (k < IN_SZ) ? W_ih[(long)j * WIH_W + k]
                          : W_hh[(long)j * HID + (k - IN_SZ)];
    wcat_bf[idx] = f2bf(v);
    return;
  }
  idx -= 4 * HID * KCAT;
  if (idx < NPAD * HID) {       // W_gen padded 100 -> 128 rows
    int r = idx / HID;
    wgen_bf[idx] = (r < NC) ? f2bf(W_gen[idx]) : (u16)0;
    return;
  }
  idx -= NPAD * HID;
  if (idx < 4 * HID) { bsum[idx] = b_ih[idx] + b_hh[idx]; return; }
  idx -= 4 * HID;
  if (idx < NPAD) { bgen_pad[idx] = (idx < NC) ? b_gen[idx] : 0.f; return; }
}

__global__ void init_state(float* __restrict__ cbuf, float* __restrict__ xcat) {
  int idx = blockIdx.x * blockDim.x + threadIdx.x;
  if (idx >= B_SZ * HID) return;
  int b = idx / HID, j = idx % HID;
  cbuf[idx] = 0.f;
  xcat[b * KCAT + IN_SZ + j] = 0.f;
}

__global__ void pack_logits(const float* __restrict__ tmp, float* __restrict__ out,
                            int total) {
  int idx = blockIdx.x * blockDim.x + threadIdx.x;
  if (idx >= total) return;
  int n = idx % NC;
  int bs = idx / NC;
  out[idx] = (n == BLANK) ? -10000.f : tmp[(long)bs * NPAD + n];
}

// ---------------------------------------------------------------------------
extern "C" void kernel_launch(void* const* d_in, const int* in_sizes, int n_in,
                              void* d_out, int out_size, void* d_ws, size_t ws_size,
                              hipStream_t stream) {
  const float* batch_H = (const float*)d_in[0];
  const int*   text    = (const int*)d_in[1];
  // d_in[2] = batch_max_length (unused; STEPS fixed)
  const float* W_i2h   = (const float*)d_in[3];
  const float* W_h2h   = (const float*)d_in[4];
  const float* b_h2h   = (const float*)d_in[5];
  const float* W_score = (const float*)d_in[6];
  const float* W_ih    = (const float*)d_in[7];
  const float* W_hh    = (const float*)d_in[8];
  const float* b_ih    = (const float*)d_in[9];
  const float* b_hh    = (const float*)d_in[10];
  const float* W_gen   = (const float*)d_in[11];
  const float* b_gen   = (const float*)d_in[12];
  float* out = (float*)d_out;

  char* p = (char*)d_ws;
  auto alloc = [&](size_t bytes) -> char* {
    char* r = p; p += (bytes + 255) & ~size_t(255); return r;
  };
  float* proj_H  = (float*)alloc((size_t)B_SZ * T_SZ * HID * 4);   // 64 MB
  float* proj_h  = (float*)alloc((size_t)B_SZ * HID * 4);
  float* xcat    = (float*)alloc((size_t)B_SZ * KCAT * 4);         // [ctx | h]
  float* cbuf    = (float*)alloc((size_t)B_SZ * HID * 4);
  float* gates   = (float*)alloc((size_t)B_SZ * 4 * HID * 4);
  float* out_hid = (float*)alloc((size_t)B_SZ * STEPS * HID * 4);
  float* ltmp    = (float*)alloc((size_t)B_SZ * STEPS * NPAD * 4);
  u16* wi2h_bf   = (u16*)alloc((size_t)HID * IN_SZ * 2);
  u16* wh2h_bf   = (u16*)alloc((size_t)HID * HID * 2);
  u16* wcat_bf   = (u16*)alloc((size_t)4 * HID * KCAT * 2);
  u16* wgen_bf   = (u16*)alloc((size_t)NPAD * HID * 2);
  float* bsum     = (float*)alloc((size_t)4 * HID * 4);
  float* bgen_pad = (float*)alloc((size_t)NPAD * 4);

  const int conv_total = HID * IN_SZ + HID * HID + 4 * HID * KCAT +
                         NPAD * HID + 4 * HID + NPAD;
  convert_weights<<<(conv_total + 255) / 256, 256, 0, stream>>>(
      W_i2h, W_h2h, W_ih, W_hh, W_gen, b_ih, b_hh, b_gen,
      wi2h_bf, wh2h_bf, wcat_bf, wgen_bf, bsum, bgen_pad, conv_total);
  init_state<<<(B_SZ * HID + 255) / 256, 256, 0, stream>>>(cbuf, xcat);

  const dim3 blk(256);
  {  // proj_H = batch_H @ W_i2h^T        (65536 x 256 x 256)
    dim3 grid(HID / 64, (B_SZ * T_SZ) / 128);
    gemm_wmma_bf16<<<grid, blk, 0, stream>>>(batch_H, IN_SZ, wi2h_bf, IN_SZ,
                                             proj_H, HID, nullptr,
                                             B_SZ * T_SZ, HID, IN_SZ);
  }

  for (int step = 0; step < STEPS; ++step) {
    {  // proj_h = h @ W_h2h^T + b_h2h    (A = h view inside xcat, lda=512)
      dim3 grid(HID / 64, B_SZ / 128);
      gemm_wmma_bf16<<<grid, blk, 0, stream>>>(xcat + IN_SZ, KCAT, wh2h_bf, HID,
                                               proj_h, HID, b_h2h,
                                               B_SZ, HID, HID);
    }
    attention_ctx<<<B_SZ, 256, 0, stream>>>(proj_H, proj_h, batch_H, W_score, xcat);
    {  // gates = [ctx|h] @ [W_ih[:,:256];W_hh]^T + (b_ih+b_hh)   (512x1024x512)
      dim3 grid((4 * HID) / 64, B_SZ / 128);
      gemm_wmma_bf16<<<grid, blk, 0, stream>>>(xcat, KCAT, wcat_bf, KCAT,
                                               gates, 4 * HID, bsum,
                                               B_SZ, 4 * HID, KCAT);
    }
    lstm_cell<<<B_SZ, 256, 0, stream>>>(gates, W_ih, text, step, cbuf, xcat, out_hid);
  }

  {  // logits = out_hid @ W_gen^T + b_gen   (13312 x 128 x 256, padded N)
    dim3 grid(NPAD / 64, (B_SZ * STEPS) / 128);
    gemm_wmma_bf16<<<grid, blk, 0, stream>>>(out_hid, HID, wgen_bf, HID,
                                             ltmp, NPAD, bgen_pad,
                                             B_SZ * STEPS, NPAD, HID);
  }
  const int total = B_SZ * STEPS * NC;
  pack_logits<<<(total + 255) / 256, 256, 0, stream>>>(ltmp, out, total);
}
